// QuantizedColumnParallel_38397007626435
// MI455X (gfx1250) — matrix operand order repaired and checked by reading, not compiled, not run on hardware
//
#include <hip/hip_runtime.h>

// QuantizedColumnParallel: y[m,n] = sum_k x[m,k] * (wq[n,k]*scale) + bias[n]
// M=4096 (B*S), K=4096, N=16384.
// Strategy: dequant int8->f16 (exact) and x f32->f16 ONCE into d_ws, then run a
// pure-f16 WMMA GEMM whose inner loop is 16 b128 loads : 16 v_wmma (no cvt VALU).

typedef _Float16 half_t;
typedef __attribute__((ext_vector_type(16))) _Float16 v16h;
typedef __attribute__((ext_vector_type(8)))  _Float16 v8h;
typedef __attribute__((ext_vector_type(8)))  float    v8f;
typedef __attribute__((ext_vector_type(4)))  float    v4f;
typedef __attribute__((ext_vector_type(4)))  int      v4i;

#define M_TOTAL 4096
#define N_TOTAL 16384
#define K_TOTAL 4096

// ---------------------------------------------------------------------------
// Prep kernels: elementwise conversion into workspace (8 elems / thread).
// ---------------------------------------------------------------------------
__global__ __launch_bounds__(256) void convert_w_kernel(
    const int* __restrict__ wq, half_t* __restrict__ w16)
{
    const size_t i = ((size_t)blockIdx.x * 256 + threadIdx.x) * 8;
    v4i a0 = *(const v4i*)(wq + i);
    v4i a1 = *(const v4i*)(wq + i + 4);
    v8h o;
#pragma unroll
    for (int e = 0; e < 4; ++e) {
        o[e]     = (half_t)a0[e];   // int8-valued: exact in f16
        o[4 + e] = (half_t)a1[e];
    }
    *(v8h*)(w16 + i) = o;
}

__global__ __launch_bounds__(256) void convert_x_kernel(
    const float* __restrict__ x, half_t* __restrict__ x16)
{
    const size_t i = ((size_t)blockIdx.x * 256 + threadIdx.x) * 8;
    v4f a0 = *(const v4f*)(x + i);
    v4f a1 = *(const v4f*)(x + i + 4);
    v8h o;
#pragma unroll
    for (int e = 0; e < 4; ++e) {
        o[e]     = (half_t)a0[e];
        o[4 + e] = (half_t)a1[e];
    }
    *(v8h*)(x16 + i) = o;
}

// ---------------------------------------------------------------------------
// Main GEMM (preconverted f16 operands).
// 8 waves per workgroup, 2(M) x 4(N); wave tile 64x64 = 4x4 WMMA tiles.
// Workgroup tile: 128(M) x 256(N).
// ---------------------------------------------------------------------------
__global__ __launch_bounds__(256) void qcp_wmma_gemm_f16(
    const half_t* __restrict__ x16,   // [M, K] row-major f16
    const half_t* __restrict__ w16,   // [N, K] row-major f16
    const float*  __restrict__ scale_p,
    const float*  __restrict__ bias,
    float*        __restrict__ y)     // [M, N] row-major
{
    const int lane  = threadIdx.x & 31;
    const int wave  = threadIdx.x >> 5;
    const int waveM = wave & 1;
    const int waveN = wave >> 1;

    const int m0 = blockIdx.x * 128 + waveM * 64;   // blockIdx.x = M block
    const int n0 = blockIdx.y * 256 + waveN * 64;   // blockIdx.y = N block

    const int lmod  = lane & 15;
    const int lhalf = lane >> 4;

    v8f acc[4][4] = {};

    // A frag (16-bit 16x32 layout): lane holds row m=lmod; elems 0..7 at
    // k = kk + 8*lhalf, elems 8..15 at k = kk + 16 + 8*lhalf (contig halves).
    // B frag (16-bit 32x16 layout): lane holds col n=lmod; same k order as A:
    // elems 0..7 at k = kk + 8*lhalf, elems 8..15 at k = kk + 16 + 8*lhalf.
    const half_t* aBase = x16 + (size_t)(m0 + lmod) * K_TOTAL + lhalf * 8;
    const half_t* bBase = w16 + (size_t)(n0 + lmod) * K_TOTAL + lhalf * 8;

    for (int kk = 0; kk < K_TOTAL; kk += 32) {
        v16h a[4];
#pragma unroll
        for (int mt = 0; mt < 4; ++mt) {
            const half_t* ap = aBase + (size_t)mt * 16 * K_TOTAL + kk;
            v8h lo = *(const v8h*)(ap);        // elems 0..7
            v8h hi = *(const v8h*)(ap + 16);   // elems 8..15
            a[mt] = __builtin_shufflevector(lo, hi,
                     0, 1, 2, 3, 4, 5, 6, 7, 8, 9, 10, 11, 12, 13, 14, 15);
        }
        v16h b[4];
#pragma unroll
        for (int nt = 0; nt < 4; ++nt) {
            const half_t* bp = bBase + (size_t)nt * 16 * K_TOTAL + kk;
            v8h blo = *(const v8h*)(bp);
            v8h bhi = *(const v8h*)(bp + 16);
            b[nt] = __builtin_shufflevector(blo, bhi,
                     0, 1, 2, 3, 4, 5, 6, 7, 8, 9, 10, 11, 12, 13, 14, 15);
        }
#pragma unroll
        for (int mt = 0; mt < 4; ++mt)
#pragma unroll
            for (int nt = 0; nt < 4; ++nt)
                acc[mt][nt] = __builtin_amdgcn_wmma_f32_16x16x32_f16(
                    false, a[mt], false, b[nt],
                    (short)0, acc[mt][nt], false, false);
    }

    const float s = scale_p[0];
#pragma unroll
    for (int nt = 0; nt < 4; ++nt) {
        const int   n  = n0 + nt * 16 + lmod;
        const float bv = bias[n];
#pragma unroll
        for (int mt = 0; mt < 4; ++mt) {
#pragma unroll
            for (int r = 0; r < 8; ++r) {
                const int m = m0 + mt * 16 + lhalf * 8 + r;
                y[(size_t)m * N_TOTAL + n] = acc[mt][nt][r] * s + bv;
            }
        }
    }
}

// ---------------------------------------------------------------------------
// Fallback: inline-convert GEMM (used only if workspace is too small).
// ---------------------------------------------------------------------------
__global__ __launch_bounds__(256) void qcp_wmma_gemm_inline(
    const float* __restrict__ x, const int* __restrict__ wq,
    const float* __restrict__ scale_p, const float* __restrict__ bias,
    float* __restrict__ y)
{
    const int lane  = threadIdx.x & 31;
    const int wave  = threadIdx.x >> 5;
    const int waveM = wave & 1;
    const int waveN = wave >> 1;
    const int m0 = blockIdx.x * 128 + waveM * 64;
    const int n0 = blockIdx.y * 256 + waveN * 64;
    const int lmod  = lane & 15;
    const int lhalf = lane >> 4;

    v8f acc[4][4] = {};
    const float* aBase = x  + (size_t)(m0 + lmod) * K_TOTAL + lhalf * 8;
    const int*   bBase = wq + (size_t)(n0 + lmod) * K_TOTAL + lhalf * 8;

    for (int kk = 0; kk < K_TOTAL; kk += 32) {
        v16h a[4];
#pragma unroll
        for (int mt = 0; mt < 4; ++mt) {
            const float* ap = aBase + (size_t)mt * 16 * K_TOTAL + kk;
            v4f f0 = *(const v4f*)(ap);
            v4f f1 = *(const v4f*)(ap + 4);
            v4f f2 = *(const v4f*)(ap + 16);
            v4f f3 = *(const v4f*)(ap + 20);
#pragma unroll
            for (int e = 0; e < 4; ++e) {
                a[mt][e]      = (half_t)f0[e];
                a[mt][4 + e]  = (half_t)f1[e];
                a[mt][8 + e]  = (half_t)f2[e];
                a[mt][12 + e] = (half_t)f3[e];
            }
        }
        v16h b[4];
#pragma unroll
        for (int nt = 0; nt < 4; ++nt) {
            const int* bp = bBase + (size_t)nt * 16 * K_TOTAL + kk;
            v4i i0 = *(const v4i*)(bp);
            v4i i1 = *(const v4i*)(bp + 4);
            v4i i2 = *(const v4i*)(bp + 16);
            v4i i3 = *(const v4i*)(bp + 20);
#pragma unroll
            for (int e = 0; e < 4; ++e) {
                b[nt][e]      = (half_t)i0[e];
                b[nt][4 + e]  = (half_t)i1[e];
                b[nt][8 + e]  = (half_t)i2[e];
                b[nt][12 + e] = (half_t)i3[e];
            }
        }
#pragma unroll
        for (int mt = 0; mt < 4; ++mt)
#pragma unroll
            for (int nt = 0; nt < 4; ++nt)
                acc[mt][nt] = __builtin_amdgcn_wmma_f32_16x16x32_f16(
                    false, a[mt], false, b[nt],
                    (short)0, acc[mt][nt], false, false);
    }

    const float s = scale_p[0];
#pragma unroll
    for (int nt = 0; nt < 4; ++nt) {
        const int   n  = n0 + nt * 16 + lmod;
        const float bv = bias[n];
#pragma unroll
        for (int mt = 0; mt < 4; ++mt)
#pragma unroll
            for (int r = 0; r < 8; ++r) {
                const int m = m0 + mt * 16 + lhalf * 8 + r;
                y[(size_t)m * N_TOTAL + n] = acc[mt][nt][r] * s + bv;
            }
    }
}

extern "C" void kernel_launch(void* const* d_in, const int* in_sizes, int n_in,
                              void* d_out, int out_size, void* d_ws, size_t ws_size,
                              hipStream_t stream) {
    const float* x     = (const float*)d_in[0];
    const int*   wq    = (const int*)  d_in[1];
    const float* scale = (const float*)d_in[2];
    const float* bias  = (const float*)d_in[3];
    float*       y     = (float*)d_out;
    (void)in_sizes; (void)n_in; (void)out_size;

    const size_t W_ELEMS = (size_t)N_TOTAL * K_TOTAL;       // 67,108,864
    const size_t X_ELEMS = (size_t)M_TOTAL * K_TOTAL;       // 16,777,216
    const size_t NEED    = (W_ELEMS + X_ELEMS) * sizeof(half_t); // 160 MB

    dim3 grid(M_TOTAL / 128, N_TOTAL / 256);                // (32, 64)

    if (ws_size >= NEED) {
        half_t* w16 = (half_t*)d_ws;
        half_t* x16 = (half_t*)d_ws + W_ELEMS;
        convert_w_kernel<<<(unsigned)(W_ELEMS / (256 * 8)), 256, 0, stream>>>(wq, w16);
        convert_x_kernel<<<(unsigned)(X_ELEMS / (256 * 8)), 256, 0, stream>>>(x, x16);
        qcp_wmma_gemm_f16<<<grid, 256, 0, stream>>>(x16, w16, scale, bias, y);
    } else {
        qcp_wmma_gemm_inline<<<grid, 256, 0, stream>>>(x, wq, scale, bias, y);
    }
}
